// BiLSTM_3410204033194
// MI455X (gfx1250) — matrix operand order repaired and checked by reading, not compiled, not run on hardware
//
#include <hip/hip_runtime.h>
#include <hip/hip_bf16.h>

typedef __attribute__((ext_vector_type(16))) _Float16 v16h;
typedef __attribute__((ext_vector_type(8)))  float    v8f;

#define TT 1024
#define BB 512
#define HH 20
#define GG 80          // 4*H gate columns
#define NTB 32         // batch tiles (B/16)
#define NTHREADS 160   // 5 wave32 -> one 16-col gate tile per wave

// LDS-only split barrier: drain DS ops, signal+wait workgroup barrier.
// (__syncthreads would also drain LOADcnt, stalling our global prefetches.)
__device__ __forceinline__ void lds_barrier() {
    asm volatile("s_wait_dscnt 0x0\n\t"
                 "s_barrier_signal -1\n\t"
                 "s_barrier_wait -1" ::: "memory");
}

// fast, clamp-free gate math: v_exp_f32 + v_rcp_f32, saturates gracefully
__device__ __forceinline__ float sig_(float x) {
    return __builtin_amdgcn_rcpf(1.0f + __expf(-x));
}
__device__ __forceinline__ float tanh_(float x) {
    return fmaf(2.0f, __builtin_amdgcn_rcpf(1.0f + __expf(-2.0f * x)), -1.0f);
}

// A-fragment position for K-index kl within a 16x32 f16 A tile (ISA 7.12.2):
// lanes 0-15 hold K in {0..7,16..23}, lanes 16-31 hold K in {8..15,24..31}.
__device__ __forceinline__ int afrag_lane(int m, int kl) { return m + (((kl >> 3) & 1) << 4); }
__device__ __forceinline__ int afrag_half(int kl)        { return (kl & 7) | (((kl >> 4) & 1) << 3); }

// Packed next-layer input layout (f16):
//  Xp0: (T, NTB, 32 lanes, 16 halves)  == A fragment 0 (K 0..31), pure x
//  Xp1: (T, NTB, 16 rows,  8 halves)   == x part of fragment 1 (K 32..39)
__device__ __forceinline__ void pack_write(_Float16* __restrict__ Yp0,
                                           _Float16* __restrict__ Yp1,
                                           size_t tb, int r, int dd, _Float16 hf) {
    if (dd < 32) Yp0[tb * 512 + afrag_lane(r, dd) * 16 + afrag_half(dd)] = hf;
    else         Yp1[tb * 128 + r * 8 + (dd - 32)] = hf;
}

// ---------------- layer 0: D=2 raw fp32 input, K=[x(0,1)|h(2..21)|pad] ----------------
__global__ __launch_bounds__(NTHREADS)
void lstm_l0_kernel(const float* __restrict__ X,       // (T,B,2)
                    const float* __restrict__ w_ih_b,  // (2,80,2)
                    const float* __restrict__ w_hh_b,  // (2,80,20)
                    const float* __restrict__ b_b,     // (2,80)
                    _Float16* __restrict__ Yp0,        // pack for layer 1
                    _Float16* __restrict__ Yp1)
{
    const int dir = blockIdx.y, btile = blockIdx.x;
    const int tid = threadIdx.x, lane = tid & 31, nt = tid >> 5;
    const float* w_ih = w_ih_b + (size_t)dir * GG * 2;
    const float* w_hh = w_hh_b + (size_t)dir * GG * HH;
    const float* bias = b_b    + (size_t)dir * GG;

    __shared__ __align__(32) _Float16 afrag[32][16];
    __shared__ float gates[16][GG];

    const int ncol = nt * 16 + (lane & 15);
    v16h bfrag;
    #pragma unroll
    for (int j = 0; j < 16; ++j) {
        const int k = ((lane >> 4) << 3) + (j & 7) + ((j & 8) << 1);
        float w = 0.0f;
        if (k < 2)            w = w_ih[ncol * 2 + k];
        else if (k < 2 + HH)  w = w_hh[ncol * HH + (k - 2)];
        bfrag[j] = (_Float16)w;
    }
    const float bcol = bias[ncol];
    v8f cinit;
    #pragma unroll
    for (int v = 0; v < 8; ++v) cinit[v] = bcol;

    for (int p = tid; p < 32 * 16; p += NTHREADS) ((_Float16*)afrag)[p] = (_Float16)0.0f;

    const int i0 = 2 * tid, i1 = 2 * tid + 1;
    const int r0 = i0 / HH, j0 = i0 % HH, r1 = i1 / HH, j1 = i1 % HH;
    float c0 = 0.0f, c1 = 0.0f;

    // x pipeline: 32 values (16 rows x 2), one per thread for tid<32
    float xreg = 0.0f;
    {
        const int t0 = dir ? (TT - 1) : 0;
        if (tid < 32)
            xreg = X[((size_t)t0 * BB + btile * 16 + (tid >> 1)) * 2 + (tid & 1)];
    }

    for (int s = 0; s < TT; ++s) {
        const int t = dir ? (TT - 1 - s) : s;

        if (tid < 32) afrag[tid >> 1][tid & 1] = (_Float16)xreg;  // kl<8: lane=m, half=kl
        lds_barrier();     // x + h(prev) visible

        {   // prefetch next step's x; consumed only at next iteration's store
            const int sn = (s + 1 < TT) ? s + 1 : s;
            const int tn = dir ? (TT - 1 - sn) : sn;
            if (tid < 32)
                xreg = X[((size_t)tn * BB + btile * 16 + (tid >> 1)) * 2 + (tid & 1)];
        }

        v8f c = cinit;
        {
            v16h a = *(const v16h*)(&afrag[lane][0]);
            c = __builtin_amdgcn_wmma_f32_16x16x32_f16(false, a, false, bfrag,
                                                       (short)0, c, false, false);
        }
        #pragma unroll
        for (int v = 0; v < 8; ++v)
            gates[v + ((lane >> 4) << 3)][ncol] = c[v];
        lds_barrier();     // gates visible; all A reads of this step done

        const size_t tb = (size_t)t * NTB + btile;
        {
            const float gi = gates[r0][j0],          gf = gates[r0][HH + j0];
            const float gc = gates[r0][2 * HH + j0], go = gates[r0][3 * HH + j0];
            c0 = sig_(gf) * c0 + sig_(gi) * tanh_(gc);
            const _Float16 hf = (_Float16)(sig_(go) * tanh_(c0));
            const int kl = 2 + j0;
            afrag[afrag_lane(r0, kl)][afrag_half(kl)] = hf;
            pack_write(Yp0, Yp1, tb, r0, dir * HH + j0, hf);
        }
        {
            const float gi = gates[r1][j1],          gf = gates[r1][HH + j1];
            const float gc = gates[r1][2 * HH + j1], go = gates[r1][3 * HH + j1];
            c1 = sig_(gf) * c1 + sig_(gi) * tanh_(gc);
            const _Float16 hf = (_Float16)(sig_(go) * tanh_(c1));
            const int kl = 2 + j1;
            afrag[afrag_lane(r1, kl)][afrag_half(kl)] = hf;
            pack_write(Yp0, Yp1, tb, r1, dir * HH + j1, hf);
        }
    }
}

// -------- layers 1/2: packed f16 input, K=[x(0..39)|h(40..59)|pad(60..63)] --------
// OUT_MODE 0: write pack for next layer; OUT_MODE 1: write fp32 FC row (batch 511)
template<int OUT_MODE>
__global__ __launch_bounds__(NTHREADS)
void lstm_mid_kernel(const _Float16* __restrict__ Xp0,  // (T,NTB,32,16)
                     const _Float16* __restrict__ Xp1,  // (T,NTB,16,8)
                     const float* __restrict__ w_ih_b,  // (2,80,40)
                     const float* __restrict__ w_hh_b,  // (2,80,20)
                     const float* __restrict__ b_b,     // (2,80)
                     _Float16* __restrict__ Yp0,
                     _Float16* __restrict__ Yp1,
                     float* __restrict__ Yfc)           // (T,40)
{
    const int dir = blockIdx.y, btile = blockIdx.x;
    const int tid = threadIdx.x, lane = tid & 31, nt = tid >> 5;
    const float* w_ih = w_ih_b + (size_t)dir * GG * 40;
    const float* w_hh = w_hh_b + (size_t)dir * GG * HH;
    const float* bias = b_b    + (size_t)dir * GG;

    __shared__ __align__(32) _Float16 afrag1[32][16];   // fragment 1 staging
    __shared__ float gates[16][GG];

    const int ncol = nt * 16 + (lane & 15);
    v16h bfrag[2];
    #pragma unroll
    for (int f = 0; f < 2; ++f) {
        #pragma unroll
        for (int j = 0; j < 16; ++j) {
            const int k = f * 32 + ((lane >> 4) << 3) + (j & 7) + ((j & 8) << 1);
            float w = 0.0f;
            if (k < 40)           w = w_ih[ncol * 40 + k];
            else if (k < 40 + HH) w = w_hh[ncol * HH + (k - 40)];
            bfrag[f][j] = (_Float16)w;
        }
    }
    const float bcol = bias[ncol];
    v8f cinit;
    #pragma unroll
    for (int v = 0; v < 8; ++v) cinit[v] = bcol;

    for (int p = tid; p < 32 * 16; p += NTHREADS) ((_Float16*)afrag1)[p] = (_Float16)0.0f;

    const int i0 = 2 * tid, i1 = 2 * tid + 1;
    const int r0 = i0 / HH, j0 = i0 % HH, r1 = i1 / HH, j1 = i1 % HH;
    float c0 = 0.0f, c1 = 0.0f;

    // pipeline regs: fragment 0 (pure x) + frag-1 x part, one step ahead
    v16h a0reg;
    uint4 x1reg = make_uint4(0, 0, 0, 0);
    {
        const int t0 = dir ? (TT - 1) : 0;
        a0reg = *(const v16h*)&Xp0[(((size_t)t0 * NTB + btile) * 32 + lane) * 16];
        if (tid < 16)
            x1reg = *(const uint4*)&Xp1[(((size_t)t0 * NTB + btile) * 16 + tid) * 8];
    }

    for (int s = 0; s < TT; ++s) {
        const int t = dir ? (TT - 1 - s) : s;

        // stage frag-1 x part (16B per row, rows 0..15)
        if (tid < 16) *(uint4*)(&afrag1[tid][0]) = x1reg;
        lds_barrier();     // frag1 x + h(prev) visible

        // prefetch next step (L2-resident packed stream; wait lands next iter)
        v16h a0n = a0reg;
        uint4 x1n = x1reg;
        {
            const int sn = (s + 1 < TT) ? s + 1 : s;
            const int tn = dir ? (TT - 1 - sn) : sn;
            a0n = *(const v16h*)&Xp0[(((size_t)tn * NTB + btile) * 32 + lane) * 16];
            if (tid < 16)
                x1n = *(const uint4*)&Xp1[(((size_t)tn * NTB + btile) * 16 + tid) * 8];
        }

        v8f c = cinit;
        c = __builtin_amdgcn_wmma_f32_16x16x32_f16(false, a0reg, false, bfrag[0],
                                                   (short)0, c, false, false);
        {
            v16h a1 = *(const v16h*)(&afrag1[lane][0]);
            c = __builtin_amdgcn_wmma_f32_16x16x32_f16(false, a1, false, bfrag[1],
                                                       (short)0, c, false, false);
        }
        #pragma unroll
        for (int v = 0; v < 8; ++v)
            gates[v + ((lane >> 4) << 3)][ncol] = c[v];
        lds_barrier();     // gates visible; all A reads of this step done

        const size_t tb = (size_t)t * NTB + btile;
        {
            const float gi = gates[r0][j0],          gf = gates[r0][HH + j0];
            const float gc = gates[r0][2 * HH + j0], go = gates[r0][3 * HH + j0];
            c0 = sig_(gf) * c0 + sig_(gi) * tanh_(gc);
            const float h = sig_(go) * tanh_(c0);
            const _Float16 hf = (_Float16)h;
            const int kl = 8 + j0;                       // K = 40+j -> kl = 8+j
            afrag1[afrag_lane(r0, kl)][afrag_half(kl)] = hf;
            if constexpr (OUT_MODE == 0) pack_write(Yp0, Yp1, tb, r0, dir * HH + j0, hf);
            else if (btile == NTB - 1 && r0 == 15) Yfc[t * 40 + dir * HH + j0] = h;
        }
        {
            const float gi = gates[r1][j1],          gf = gates[r1][HH + j1];
            const float gc = gates[r1][2 * HH + j1], go = gates[r1][3 * HH + j1];
            c1 = sig_(gf) * c1 + sig_(gi) * tanh_(gc);
            const float h = sig_(go) * tanh_(c1);
            const _Float16 hf = (_Float16)h;
            const int kl = 8 + j1;
            afrag1[afrag_lane(r1, kl)][afrag_half(kl)] = hf;
            if constexpr (OUT_MODE == 0) pack_write(Yp0, Yp1, tb, r1, dir * HH + j1, hf);
            else if (btile == NTB - 1 && r1 == 15) Yfc[t * 40 + dir * HH + j1] = h;
        }
        a0reg = a0n;
        x1reg = x1n;
    }
}

// out[t,:] = Yfc[t,:] @ fc_w^T + fc_b   (T=1024, NC=4)
__global__ void fc_kernel(const float* __restrict__ Yfc,
                          const float* __restrict__ fc_w,
                          const float* __restrict__ fc_b,
                          float* __restrict__ out)
{
    const int t = blockIdx.x * blockDim.x + threadIdx.x;
    if (t >= TT) return;
    const float* row = Yfc + t * 40;
    #pragma unroll
    for (int c = 0; c < 4; ++c) {
        float acc = fc_b[c];
        #pragma unroll
        for (int k = 0; k < 40; ++k) acc += row[k] * fc_w[c * 40 + k];
        out[t * 4 + c] = acc;
    }
}

extern "C" void kernel_launch(void* const* d_in, const int* in_sizes, int n_in,
                              void* d_out, int out_size, void* d_ws, size_t ws_size,
                              hipStream_t stream)
{
    (void)in_sizes; (void)n_in; (void)out_size; (void)ws_size;
    const float* x      = (const float*)d_in[0];
    const float* w_ih0  = (const float*)d_in[1];
    const float* w_hh0  = (const float*)d_in[2];
    const float* b0     = (const float*)d_in[3];
    const float* w_ih12 = (const float*)d_in[4];
    const float* w_hh12 = (const float*)d_in[5];
    const float* b12    = (const float*)d_in[6];
    const float* fc_w   = (const float*)d_in[7];
    const float* fc_b   = (const float*)d_in[8];
    float* out = (float*)d_out;

    // workspace: two generations of packed layer inputs + FC row
    _Float16* Xp0a = (_Float16*)d_ws;                      // (T,NTB,32,16) = 32 MB
    _Float16* Xp1a = Xp0a + (size_t)TT * NTB * 512;        // (T,NTB,16,8)  =  8 MB
    _Float16* Xp0b = Xp1a + (size_t)TT * NTB * 128;
    _Float16* Xp1b = Xp0b + (size_t)TT * NTB * 512;
    float*    Yfc  = (float*)(Xp1b + (size_t)TT * NTB * 128);  // (T,40)

    dim3 grid(NTB, 2), block(NTHREADS);
    lstm_l0_kernel<<<grid, block, 0, stream>>>(x, w_ih0, w_hh0, b0, Xp0a, Xp1a);
    lstm_mid_kernel<0><<<grid, block, 0, stream>>>(Xp0a, Xp1a, w_ih12, w_hh12, b12,
                                                   Xp0b, Xp1b, nullptr);
    lstm_mid_kernel<1><<<grid, block, 0, stream>>>(
        Xp0b, Xp1b,
        w_ih12 + (size_t)2 * GG * 40, w_hh12 + (size_t)2 * GG * HH, b12 + 2 * GG,
        nullptr, nullptr, Yfc);
    fc_kernel<<<dim3(8), dim3(128), 0, stream>>>(Yfc, fc_w, fc_b, out);
}
